// LinearTransformerPredictor_50878182588747
// MI455X (gfx1250) — matrix operand order repaired and checked by reading, not compiled, not run on hardware
//
#include <hip/hip_runtime.h>
#include <cstdint>
#include <cstddef>

// Problem constants (match reference)
#define B_      2
#define T_      2048
#define DM      512
#define NH      8
#define DH      64
#define DFF     2048
#define M_ROWS  (B_ * T_)   // 4096 token rows

typedef __attribute__((ext_vector_type(16))) _Float16 v16h;
typedef __attribute__((ext_vector_type(8)))  _Float16 v8h;
typedef __attribute__((ext_vector_type(8)))  float    v8f;

// ---------------------------------------------------------------------------
// CDNA5 async global->LDS copy (ASYNCcnt-tracked) + wait, via inline asm.
// LDS aperture is 4GB-aligned: low 32 bits of a flat shared pointer are the
// workgroup-relative LDS byte offset, which is what VDST wants (ISA 10.2).
// ---------------------------------------------------------------------------
__device__ __forceinline__ unsigned lds_off32(const void* p)
{
    return (unsigned)(unsigned long long)(uintptr_t)p;
}
__device__ __forceinline__ void async_ld_b32(unsigned lds_off, const float* gp)
{
    asm volatile("global_load_async_to_lds_b32 %0, %1, off"
                 :: "v"(lds_off), "v"((unsigned long long)(uintptr_t)gp)
                 : "memory");
}

// ---------------------------------------------------------------------------
// LayerNorm over last dim (512), writes f16 activations for WMMA A-operand.
// One 256-thread block per row; each thread handles 2 elements.
// ---------------------------------------------------------------------------
__global__ void ln_f16_kernel(const float* __restrict__ x,
                              const float* __restrict__ gamma,
                              const float* __restrict__ beta,
                              _Float16* __restrict__ out)
{
    __shared__ float s1[256];
    __shared__ float s2[256];
    const int tid = threadIdx.x;
    const size_t row = blockIdx.x;
    const float* xr = x + row * DM;
    const float a = xr[tid];
    const float b = xr[tid + 256];
    s1[tid] = a + b;
    s2[tid] = a * a + b * b;
    __syncthreads();
    for (int s = 128; s > 0; s >>= 1) {
        if (tid < s) { s1[tid] += s1[tid + s]; s2[tid] += s2[tid + s]; }
        __syncthreads();
    }
    const float mu  = s1[0] * (1.0f / DM);
    const float var = s2[0] * (1.0f / DM) - mu * mu;
    const float rs  = rsqrtf(var + 1e-5f);
    out[row * DM + tid]       = (_Float16)((a - mu) * rs * gamma[tid]       + beta[tid]);
    out[row * DM + tid + 256] = (_Float16)((b - mu) * rs * gamma[tid + 256] + beta[tid + 256]);
}

// ---------------------------------------------------------------------------
// Weight conversion: W (K x N, f32 row-major) -> Wt (N x K, f16).
// Transposed layout makes the WMMA B-fragment a contiguous per-lane load.
// ---------------------------------------------------------------------------
__global__ void wt_f16_kernel(const float* __restrict__ W,
                              _Float16* __restrict__ Wt,
                              int K, int N)
{
    const size_t idx = (size_t)blockIdx.x * blockDim.x + threadIdx.x;
    if (idx >= (size_t)K * N) return;
    const int n = (int)(idx / K);
    const int k = (int)(idx % K);
    Wt[idx] = (_Float16)W[(size_t)k * N + n];
}

// ---------------------------------------------------------------------------
// Epilogue helper: D tile layout -> VGPR i holds row (mbase + i + 8g), col n.
// ---------------------------------------------------------------------------
template <int ACT, bool HOUT>
__device__ __forceinline__ void store_tile(const v8f& c, int mbase, int g, int n,
                                           int N, float bn,
                                           const float* __restrict__ resid,
                                           float* __restrict__ Cf,
                                           _Float16* __restrict__ Ch)
{
#pragma unroll
    for (int i = 0; i < 8; ++i) {
        const int m = mbase + i + 8 * g;
        float v = c[i] + bn;
        if (ACT == 1) v = (v > 0.0f) ? (v + 1.0f) : __expf(v);   // elu(z)+1
        else if (ACT == 2) v = fmaxf(v, 0.0f);                   // relu
        if (resid) v += resid[(size_t)m * N + n];
        if (HOUT) Ch[(size_t)m * N + n] = (_Float16)v;
        else      Cf[(size_t)m * N + n] = v;
    }
}

__device__ __forceinline__ v16h frag_cat(const v8h lo, const v8h hi)
{
    return __builtin_shufflevector(lo, hi, 0,1,2,3,4,5,6,7,8,9,10,11,12,13,14,15);
}

// ---------------------------------------------------------------------------
// WMMA GEMM, register-tiled: block = 128 threads = 4 waves arranged 2x2,
// block tile 64(M) x 64(N); each wave computes a 32x32 tile with 2x2
// v8f accumulators, so every A/B fragment feeds two WMMAs (AI ~16 f/B).
//   A   : f16 row-major (token activations), M x K
//   Bt  : f16, B transposed -> [N][K]
//   ACT : 0 = none, 1 = elu(z)+1, 2 = relu;  HOUT: store f16 vs f32
// ---------------------------------------------------------------------------
template <int ACT, bool HOUT>
__global__ void gemm_wmma_kernel(const _Float16* __restrict__ A,
                                 const _Float16* __restrict__ Bt,
                                 const float* __restrict__ bias,
                                 const float* __restrict__ resid,
                                 float* __restrict__ Cf,
                                 _Float16* __restrict__ Ch,
                                 int N, int K)
{
    const int lane = threadIdx.x & 31;   // wave32
    const int wave = threadIdx.x >> 5;   // 0..3
    const int g    = lane >> 4;          // half-wave group
    const int l    = lane & 15;
    const int mw   = wave & 1;
    const int nw   = wave >> 1;
    const int mbase = blockIdx.y * 64 + mw * 32;
    const int nbase = blockIdx.x * 64 + nw * 32;

    // A fragment (16x32 f16): lane row M=l; halves 0..7 = K[8g..], 8..15 = K[16+8g..]
    const _Float16* ap0 = A + (size_t)(mbase + l) * K + 8 * g;
    const _Float16* ap1 = ap0 + (size_t)16 * K;
    // B fragment (32x16): lane column N=l; halves 0..7 = K[8g..], 8..15 = K[16+8g..]
    const _Float16* bp0 = Bt + (size_t)(nbase + l) * K + 8 * g;
    const _Float16* bp1 = bp0 + (size_t)16 * K;

    v8f c00 = {}, c01 = {}, c10 = {}, c11 = {};
    for (int k0 = 0; k0 < K; k0 += 32) {
        const v16h a0 = frag_cat(*(const v8h*)(ap0 + k0), *(const v8h*)(ap0 + k0 + 16));
        const v16h a1 = frag_cat(*(const v8h*)(ap1 + k0), *(const v8h*)(ap1 + k0 + 16));
        const v16h b0 = frag_cat(*(const v8h*)(bp0 + k0), *(const v8h*)(bp0 + k0 + 16));
        const v16h b1 = frag_cat(*(const v8h*)(bp1 + k0), *(const v8h*)(bp1 + k0 + 16));
        c00 = __builtin_amdgcn_wmma_f32_16x16x32_f16(false, a0, false, b0, (short)0, c00, false, false);
        c01 = __builtin_amdgcn_wmma_f32_16x16x32_f16(false, a0, false, b1, (short)0, c01, false, false);
        c10 = __builtin_amdgcn_wmma_f32_16x16x32_f16(false, a1, false, b0, (short)0, c10, false, false);
        c11 = __builtin_amdgcn_wmma_f32_16x16x32_f16(false, a1, false, b1, (short)0, c11, false, false);
    }

    const int n0 = nbase + l;
    const int n1 = nbase + 16 + l;
    const float bn0 = bias ? bias[n0] : 0.0f;
    const float bn1 = bias ? bias[n1] : 0.0f;
    store_tile<ACT, HOUT>(c00, mbase,      g, n0, N, bn0, resid, Cf, Ch);
    store_tile<ACT, HOUT>(c01, mbase,      g, n1, N, bn1, resid, Cf, Ch);
    store_tile<ACT, HOUT>(c10, mbase + 16, g, n0, N, bn0, resid, Cf, Ch);
    store_tile<ACT, HOUT>(c11, mbase + 16, g, n1, N, bn1, resid, Cf, Ch);
}

// ---------------------------------------------------------------------------
// Causal linear attention scan. One 64-thread block (2 waves) per (b,h).
// Thread m owns state column S[:, m] (64 f32 in registers, fully unrolled).
// Q/K/V rows for step t+1 are prefetched into the alternate LDS buffer with
// CDNA5 async global->LDS copies while step t computes, hiding the global
// latency inside the inherently serial T recurrence.
// ---------------------------------------------------------------------------
__global__ void lin_attn_kernel(const float* __restrict__ Qb,
                                const float* __restrict__ Kb,
                                const float* __restrict__ Vb,
                                _Float16* __restrict__ out)
{
    const int tid = threadIdx.x;           // output column m within head
    const int b   = blockIdx.x >> 3;
    const int h   = blockIdx.x & 7;
    const size_t base = (size_t)b * T_ * DM + (size_t)h * DH;

    __shared__ float sK[2][DH];
    __shared__ float sQ[2][DH];
    __shared__ float sV[2][DH];
    __shared__ float sZ[DH];
    sZ[tid] = 0.0f;

    float S[DH];
#pragma unroll
    for (int d = 0; d < DH; ++d) S[d] = 0.0f;

    // Prologue: stage t=0 into buffer 0 (3 async b32 copies per lane).
    {
        const size_t r0 = base + tid;
        async_ld_b32(lds_off32(&sK[0][tid]), Kb + r0);
        async_ld_b32(lds_off32(&sQ[0][tid]), Qb + r0);
        async_ld_b32(lds_off32(&sV[0][tid]), Vb + r0);
    }

    for (int t = 0; t < T_; ++t) {
        const int cur = t & 1;
        if (t + 1 < T_) {
            // Prefetch next step into the other buffer, then wait for the
            // 3 outstanding copies belonging to the *current* step.
            const size_t rn = base + (size_t)(t + 1) * DM + tid;
            async_ld_b32(lds_off32(&sK[cur ^ 1][tid]), Kb + rn);
            async_ld_b32(lds_off32(&sQ[cur ^ 1][tid]), Qb + rn);
            async_ld_b32(lds_off32(&sV[cur ^ 1][tid]), Vb + rn);
            asm volatile("s_wait_asynccnt 0x3" ::: "memory");
        } else {
            asm volatile("s_wait_asynccnt 0x0" ::: "memory");
        }
        __syncthreads();

        sZ[tid] += sK[cur][tid];            // z_t = cumsum(K), inclusive
        const float vm = sV[cur][tid];
        __syncthreads();

        float acc = 0.0f, den = 0.0f;
#pragma unroll
        for (int d = 0; d < DH; ++d) {
            S[d] = fmaf(sK[cur][d], vm, S[d]);   // KV_t inclusive update
            acc  = fmaf(sQ[cur][d], S[d], acc);  // num_m = q . S[:,m]
            den  = fmaf(sQ[cur][d], sZ[d], den); // den   = q . z
        }
        den = fmaxf(den, 1e-6f);
        out[base + (size_t)t * DM + tid] = (_Float16)(acc / den);
        __syncthreads();                    // protect LDS buffers for next step
    }
}

// ---------------------------------------------------------------------------
extern "C" void kernel_launch(void* const* d_in, const int* in_sizes, int n_in,
                              void* d_out, int out_size, void* d_ws, size_t ws_size,
                              hipStream_t stream)
{
    (void)in_sizes; (void)n_in; (void)out_size; (void)ws_size;

    const float* x    = (const float*)d_in[0];
    const float* ln1g = (const float*)d_in[1];
    const float* ln1b = (const float*)d_in[2];
    const float* Wq   = (const float*)d_in[3];
    const float* bq   = (const float*)d_in[4];
    const float* Wk   = (const float*)d_in[5];
    const float* bk   = (const float*)d_in[6];
    const float* Wv   = (const float*)d_in[7];
    const float* bv   = (const float*)d_in[8];
    const float* Wo   = (const float*)d_in[9];
    const float* bo   = (const float*)d_in[10];
    const float* ln2g = (const float*)d_in[11];
    const float* ln2b = (const float*)d_in[12];
    const float* W1   = (const float*)d_in[13];
    const float* b1   = (const float*)d_in[14];
    const float* W2   = (const float*)d_in[15];
    const float* b2   = (const float*)d_in[16];
    float* outp = (float*)d_out;

    // Workspace carve (256B aligned)
    uintptr_t p = (uintptr_t)d_ws;
    auto carve = [&p](size_t bytes) -> void* {
        p = (p + 255) & ~(uintptr_t)255;
        void* r = (void*)p;
        p += bytes;
        return r;
    };
    _Float16* h16    = (_Float16*)carve((size_t)M_ROWS * DM  * sizeof(_Float16));
    float*    qbuf   = (float*)   carve((size_t)M_ROWS * DM  * sizeof(float));
    float*    kbuf   = (float*)   carve((size_t)M_ROWS * DM  * sizeof(float));
    float*    vbuf   = (float*)   carve((size_t)M_ROWS * DM  * sizeof(float));
    _Float16* attn16 = (_Float16*)carve((size_t)M_ROWS * DM  * sizeof(_Float16));
    float*    x1     = (float*)   carve((size_t)M_ROWS * DM  * sizeof(float));
    _Float16* h2_16  = (_Float16*)carve((size_t)M_ROWS * DM  * sizeof(_Float16));
    _Float16* ffn16  = (_Float16*)carve((size_t)M_ROWS * DFF * sizeof(_Float16));
    _Float16* Wq_t   = (_Float16*)carve((size_t)DM  * DM  * sizeof(_Float16));
    _Float16* Wk_t   = (_Float16*)carve((size_t)DM  * DM  * sizeof(_Float16));
    _Float16* Wv_t   = (_Float16*)carve((size_t)DM  * DM  * sizeof(_Float16));
    _Float16* Wo_t   = (_Float16*)carve((size_t)DM  * DM  * sizeof(_Float16));
    _Float16* W1_t   = (_Float16*)carve((size_t)DM  * DFF * sizeof(_Float16));
    _Float16* W2_t   = (_Float16*)carve((size_t)DFF * DM  * sizeof(_Float16));

    // --- weight conversion (f32 -> f16, transposed) ---
    {
        const int thr = 256;
        const int nSq = (DM * DM + thr - 1) / thr;
        const int nFf = (DM * DFF + thr - 1) / thr;
        hipLaunchKernelGGL(wt_f16_kernel, dim3(nSq), dim3(thr), 0, stream, Wq, Wq_t, DM, DM);
        hipLaunchKernelGGL(wt_f16_kernel, dim3(nSq), dim3(thr), 0, stream, Wk, Wk_t, DM, DM);
        hipLaunchKernelGGL(wt_f16_kernel, dim3(nSq), dim3(thr), 0, stream, Wv, Wv_t, DM, DM);
        hipLaunchKernelGGL(wt_f16_kernel, dim3(nSq), dim3(thr), 0, stream, Wo, Wo_t, DM, DM);
        hipLaunchKernelGGL(wt_f16_kernel, dim3(nFf), dim3(thr), 0, stream, W1, W1_t, DM, DFF);
        hipLaunchKernelGGL(wt_f16_kernel, dim3(nFf), dim3(thr), 0, stream, W2, W2_t, DFF, DM);
    }

    // --- sublayer 1: LN1 -> QKV GEMMs -> linear attention -> Wo (+residual) ---
    hipLaunchKernelGGL(ln_f16_kernel, dim3(M_ROWS), dim3(256), 0, stream,
                       x, ln1g, ln1b, h16);

    const dim3 gblk(128);
    const dim3 ggrid_dm(DM / 64, M_ROWS / 64);            // (8, 64)
    const dim3 ggrid_ff(DFF / 64, M_ROWS / 64);           // (32, 64)

    hipLaunchKernelGGL((gemm_wmma_kernel<1, false>), ggrid_dm, gblk, 0, stream,
                       h16, Wq_t, bq, (const float*)nullptr, qbuf, (_Float16*)nullptr, DM, DM);
    hipLaunchKernelGGL((gemm_wmma_kernel<1, false>), ggrid_dm, gblk, 0, stream,
                       h16, Wk_t, bk, (const float*)nullptr, kbuf, (_Float16*)nullptr, DM, DM);
    hipLaunchKernelGGL((gemm_wmma_kernel<0, false>), ggrid_dm, gblk, 0, stream,
                       h16, Wv_t, bv, (const float*)nullptr, vbuf, (_Float16*)nullptr, DM, DM);

    hipLaunchKernelGGL(lin_attn_kernel, dim3(B_ * NH), dim3(DH), 0, stream,
                       qbuf, kbuf, vbuf, attn16);

    hipLaunchKernelGGL((gemm_wmma_kernel<0, false>), ggrid_dm, gblk, 0, stream,
                       attn16, Wo_t, bo, x, x1, (_Float16*)nullptr, DM, DM);

    // --- sublayer 2: LN2 -> FFN (relu) -> W2 (+residual) -> d_out ---
    hipLaunchKernelGGL(ln_f16_kernel, dim3(M_ROWS), dim3(256), 0, stream,
                       x1, ln2g, ln2b, h2_16);

    hipLaunchKernelGGL((gemm_wmma_kernel<2, true>), ggrid_ff, gblk, 0, stream,
                       h2_16, W1_t, b1, (const float*)nullptr, (float*)nullptr, ffn16, DFF, DM);

    hipLaunchKernelGGL((gemm_wmma_kernel<0, false>), ggrid_dm, gblk, 0, stream,
                       ffn16, W2_t, b2, x1, outp, (_Float16*)nullptr, DM, DFF);
}
